// RankingLossFunc_17437567222534
// MI455X (gfx1250) — compile-verified, matrix operand order checked by reading, not run on hardware
//
#include <hip/hip_runtime.h>

#define NROW 8192
#define DIM  256
#define DELTA 0.2f
#define EPS 1e-8f

typedef __attribute__((ext_vector_type(8)))  __bf16 v8bf;
typedef __attribute__((ext_vector_type(16))) __bf16 v16bf;
typedef __attribute__((ext_vector_type(8)))  float  v8f;

__device__ __forceinline__ __bf16 f32_to_bf16_rne(float f) {
    unsigned u = __builtin_bit_cast(unsigned, f);
    unsigned r = (u + 0x7FFFu + ((u >> 16) & 1u)) >> 16;
    unsigned short s = (unsigned short)r;
    return __builtin_bit_cast(__bf16, s);
}
__device__ __forceinline__ float bf16_to_f32(__bf16 h) {
    unsigned short s = __builtin_bit_cast(unsigned short, h);
    unsigned u = ((unsigned)s) << 16;
    return __builtin_bit_cast(float, u);
}

// ---------------------------------------------------------------------------
// Kernel 1: row normalization (eps-clamped), bf16 hi/lo split, pos = cos(X_i,Y_i)
// One 256-thread block per row (DIM == 256).
// ---------------------------------------------------------------------------
__global__ void normalize_split(const float* __restrict__ X, const float* __restrict__ Y,
                                __bf16* __restrict__ Xhi, __bf16* __restrict__ Xlo,
                                __bf16* __restrict__ Yhi, __bf16* __restrict__ Ylo,
                                float* __restrict__ pos) {
    const int row = blockIdx.x;
    const int t   = threadIdx.x;
    const float x = X[row * DIM + t];
    const float y = Y[row * DIM + t];

    __shared__ float sx[256], sy[256], sxy[256];
    sx[t] = x * x; sy[t] = y * y; sxy[t] = x * y;
    __syncthreads();
    #pragma unroll
    for (int off = 128; off > 0; off >>= 1) {
        if (t < off) { sx[t] += sx[t+off]; sy[t] += sy[t+off]; sxy[t] += sxy[t+off]; }
        __syncthreads();
    }
    __shared__ float s_ix, s_iy;
    if (t == 0) {
        float ix = 1.0f / fmaxf(sqrtf(sx[0]), EPS);
        float iy = 1.0f / fmaxf(sqrtf(sy[0]), EPS);
        s_ix = ix; s_iy = iy;
        pos[row] = sxy[0] * ix * iy;
    }
    __syncthreads();

    float xn = x * s_ix;
    float yn = y * s_iy;
    __bf16 xh = f32_to_bf16_rne(xn);
    __bf16 yh = f32_to_bf16_rne(yn);
    Xhi[row * DIM + t] = xh;
    Xlo[row * DIM + t] = f32_to_bf16_rne(xn - bf16_to_f32(xh));
    Yhi[row * DIM + t] = yh;
    Ylo[row * DIM + t] = f32_to_bf16_rne(yn - bf16_to_f32(yh));
}

// ---------------------------------------------------------------------------
// Fragment load for 16-bit A-matrix 16x32 (also valid for B of A*B^T from
// row-major data): lane L<16 holds row M=L, K = {k0..k0+7, k0+16..k0+23};
// lane L>=16 holds row M=L-16, K shifted by +8. Two b128 loads.
// ---------------------------------------------------------------------------
__device__ __forceinline__ v16bf load_frag(const __bf16* __restrict__ base,
                                           int row0, int k0, int lane) {
    const int r = row0 + (lane & 15);
    const int k = k0 + ((lane >> 4) << 3);
    const __bf16* p = base + r * DIM + k;
    union { v16bf v; v8bf h[2]; } u;
    u.h[0] = *(const v8bf*)(p);
    u.h[1] = *(const v8bf*)(p + 16);
    return u.v;
}

// ---------------------------------------------------------------------------
// Kernel 2: S tile (128x128 per block) via bf16x3 WMMA, fused hinge + reduce.
// 8 waves/block, each wave owns a 32x64 sub-tile (2x4 WMMA C tiles).
// ---------------------------------------------------------------------------
__global__ void __launch_bounds__(256)
gemm_hinge(const __bf16* __restrict__ Xhi, const __bf16* __restrict__ Xlo,
           const __bf16* __restrict__ Yhi, const __bf16* __restrict__ Ylo,
           const float* __restrict__ pos, float* __restrict__ partials) {
    const int tid  = threadIdx.x;
    const int lane = tid & 31;
    const int wave = tid >> 5;
    const int wm   = wave >> 1;   // 0..3
    const int wn   = wave & 1;    // 0..1

    const int m0 = blockIdx.y * 128 + wm * 32;  // 32 rows for this wave
    const int n0 = blockIdx.x * 128 + wn * 64;  // 64 cols for this wave

    v8f acc[2][4];
    #pragma unroll
    for (int i = 0; i < 2; ++i)
        #pragma unroll
        for (int j = 0; j < 4; ++j)
            acc[i][j] = (v8f){0.f,0.f,0.f,0.f,0.f,0.f,0.f,0.f};

    for (int k0 = 0; k0 < DIM; k0 += 32) {
        v16bf ah[2], al[2], bh[4], bl[4];
        #pragma unroll
        for (int i = 0; i < 2; ++i) {
            ah[i] = load_frag(Xhi, m0 + i * 16, k0, lane);
            al[i] = load_frag(Xlo, m0 + i * 16, k0, lane);
        }
        #pragma unroll
        for (int j = 0; j < 4; ++j) {
            bh[j] = load_frag(Yhi, n0 + j * 16, k0, lane);
            bl[j] = load_frag(Ylo, n0 + j * 16, k0, lane);
        }
        #pragma unroll
        for (int i = 0; i < 2; ++i) {
            #pragma unroll
            for (int j = 0; j < 4; ++j) {
                // bf16x3 fp32 emulation: Ah*Bh + Ah*Bl + Al*Bh (f32 accumulate)
                acc[i][j] = __builtin_amdgcn_wmma_f32_16x16x32_bf16(
                    false, ah[i], false, bh[j], (short)0, acc[i][j], false, false);
                acc[i][j] = __builtin_amdgcn_wmma_f32_16x16x32_bf16(
                    false, ah[i], false, bl[j], (short)0, acc[i][j], false, false);
                acc[i][j] = __builtin_amdgcn_wmma_f32_16x16x32_bf16(
                    false, al[i], false, bh[j], (short)0, acc[i][j], false, false);
            }
        }
    }

    // Epilogue: hinge = max(0, DELTA - pos[m] + s), diagonal masked.
    // C/D layout: VGPR r, lane L -> M = r + 8*(L>=16), N = L & 15 (within tile).
    const int lhalf = lane >> 4;
    const int nlane = lane & 15;
    float lsum = 0.0f;
    #pragma unroll
    for (int i = 0; i < 2; ++i) {
        #pragma unroll
        for (int r = 0; r < 8; ++r) {
            const int m = m0 + i * 16 + r + 8 * lhalf;
            const float base = DELTA - pos[m];
            #pragma unroll
            for (int j = 0; j < 4; ++j) {
                const int n = n0 + j * 16 + nlane;
                float h = base + acc[i][j][r];
                h = fmaxf(h, 0.0f);
                lsum += (m != n) ? h : 0.0f;
            }
        }
    }

    __shared__ float red[256];
    red[tid] = lsum;
    __syncthreads();
    #pragma unroll
    for (int off = 128; off > 0; off >>= 1) {
        if (tid < off) red[tid] += red[tid + off];
        __syncthreads();
    }
    if (tid == 0) partials[blockIdx.y * gridDim.x + blockIdx.x] = red[0];
}

// ---------------------------------------------------------------------------
// Kernel 3: deterministic final reduction of per-block partials.
// ---------------------------------------------------------------------------
__global__ void reduce_partials(const float* __restrict__ partials, float* __restrict__ out, int n) {
    __shared__ float red[256];
    float s = 0.0f;
    for (int i = threadIdx.x; i < n; i += 256) s += partials[i];
    red[threadIdx.x] = s;
    __syncthreads();
    #pragma unroll
    for (int off = 128; off > 0; off >>= 1) {
        if (threadIdx.x < off) red[threadIdx.x] += red[threadIdx.x + off];
        __syncthreads();
    }
    if (threadIdx.x == 0) out[0] = red[0];
}

extern "C" void kernel_launch(void* const* d_in, const int* in_sizes, int n_in,
                              void* d_out, int out_size, void* d_ws, size_t ws_size,
                              hipStream_t stream) {
    (void)in_sizes; (void)n_in; (void)out_size; (void)ws_size;
    const float* X = (const float*)d_in[0];
    const float* Y = (const float*)d_in[1];
    float* out = (float*)d_out;

    const size_t ND = (size_t)NROW * DIM;       // 2,097,152 elements
    char* ws = (char*)d_ws;
    __bf16* Xhi = (__bf16*)(ws);
    __bf16* Xlo = (__bf16*)(ws + 2 * ND);        // each split = 4 MiB
    __bf16* Yhi = (__bf16*)(ws + 4 * ND);
    __bf16* Ylo = (__bf16*)(ws + 6 * ND);
    float*  pos = (float*)(ws + 8 * ND);         // 32 KiB
    float*  partials = (float*)(ws + 8 * ND + sizeof(float) * NROW); // 16 KiB

    normalize_split<<<NROW, 256, 0, stream>>>(X, Y, Xhi, Xlo, Yhi, Ylo, pos);

    dim3 grid(NROW / 128, NROW / 128);           // 64 x 64 tiles
    gemm_hinge<<<grid, 256, 0, stream>>>(Xhi, Xlo, Yhi, Ylo, pos, partials);

    reduce_partials<<<1, 256, 0, stream>>>(partials, out, (NROW / 128) * (NROW / 128));
}
